// gnn_model_59425167507663
// MI455X (gfx1250) — compile-verified, matrix operand order checked by reading, not run on hardware
//
#include <hip/hip_runtime.h>
#include <math.h>

// ---------------- types ----------------
typedef _Float16 f16;
typedef _Float16 h8v  __attribute__((ext_vector_type(8)));
typedef _Float16 h16v __attribute__((ext_vector_type(16)));
typedef float    f8v  __attribute__((ext_vector_type(8)));
typedef int      v4i  __attribute__((ext_vector_type(4)));

#define CUTOFF_F 5.0f
#define NS 7
#define NR 6

#if defined(__gfx1250__) && __has_builtin(__builtin_amdgcn_global_load_async_to_lds_b128)
#define HAVE_ASYNC_LDS 1
typedef __attribute__((address_space(1))) v4i gv4i_t;
typedef __attribute__((address_space(3))) v4i lv4i_t;
#else
#define HAVE_ASYNC_LDS 0
#endif

__device__ __forceinline__ void wait_asynccnt0() {
#if defined(__gfx1250__)
#if __has_builtin(__builtin_amdgcn_s_wait_asynccnt)
    __builtin_amdgcn_s_wait_asynccnt(0);
#else
    asm volatile("s_wait_asynccnt 0" ::: "memory");
#endif
#endif
}

__device__ __forceinline__ float silu_f(float v) {
    return v / (1.0f + __expf(-v));
}

__device__ __forceinline__ float envelope_f(float x) {
    // P_ENV = 6 : 1/x - 28 x^5 + 48 x^6 - 21 x^7, gated by x<1
    float x2 = x * x;
    float x5 = x2 * x2 * x;
    float v = 1.0f / x + (-28.0f) * x5 + 48.0f * x5 * x + (-21.0f) * x5 * x2;
    return (x < 1.0f) ? v : 0.0f;
}

// A/B fragment loader per CDNA5 16-bit 16x32 layout:
// lane l: base0 = (l>>4)*8; halves[0..7] = row[k0+base0 .. +7], halves[8..15] = row[k0+base0+16 .. +23]
__device__ __forceinline__ h16v frag_load(const f16* __restrict__ rowk0, int base0) {
    h8v lo = *(const h8v*)(rowk0 + base0);
    h8v hi = *(const h8v*)(rowk0 + base0 + 16);
    h16v f;
#pragma unroll
    for (int i = 0; i < 8; ++i) { f[i] = lo[i]; f[i + 8] = hi[i]; }
    return f;
}

// ---------------- LDS-staged multi-wave WMMA GEMM ----------------
// Block = 128 threads (4 waves). Block computes a 64(M) x NT*16(N) tile of
// Y = epi(A[M,K] @ Wt[N,K]^T). Wt tile staged in LDS (shared by 4 waves);
// A fragments register-resident, reused across NT column tiles.
// flags: 1=+bias 2=silu 4=*(rbf@R12) 8=+addsrc 16=accumulate into outF (f32)
template <int NT, int K>
__global__ __launch_bounds__(128) void gemm_lds_kernel(
    const f16* __restrict__ A, const f16* __restrict__ Wt,
    const float* __restrict__ bias, const f16* __restrict__ addsrc,
    const float* __restrict__ rbf6, const float* __restrict__ R12,
    f16* __restrict__ outH, float* __restrict__ outF,
    int M, int N, int flags)
{
    constexpr int NCH  = NT * 16;
    constexpr int LDSK = K + 8;              // pad 8 halves -> bank-conflict-free B frags
    constexpr int KCH  = (K < 128) ? K : 128;
    constexpr int KS   = KCH / 32;
    __shared__ f16 sW[NCH * LDSK];

    const int nblocks = N / NCH;
    const int nblk = blockIdx.x % nblocks;
    const int mblk = blockIdx.x / nblocks;
    const int n0 = nblk * NCH;
    const int m0 = mblk * 64;

    // ---- cooperative stage of Wt[n0 .. n0+NCH) rows into LDS ----
    {
        constexpr int chunksPerRow = K >> 3;       // 16B chunks
        constexpr int total = NCH * chunksPerRow;  // multiple of 128
        for (int c = threadIdx.x; c < total; c += 128) {
            const int n = c / chunksPerRow;
            const int kc = (c % chunksPerRow) << 3;
            const f16* src = Wt + (size_t)(n0 + n) * K + kc;
            f16* dst = &sW[n * LDSK + kc];
#if HAVE_ASYNC_LDS
            __builtin_amdgcn_global_load_async_to_lds_b128(
                (gv4i_t*)src, (lv4i_t*)dst, 0, 0);
#else
            *(h8v*)dst = *(const h8v*)src;
#endif
        }
#if HAVE_ASYNC_LDS
        wait_asynccnt0();
#endif
        __syncthreads();
    }

    const int wv = threadIdx.x >> 5;
    const int lane = threadIdx.x & 31;
    const int lm = lane & 15;
    const int hi = lane >> 4;
    const int base0 = hi * 8;

    int mrow = m0 + wv * 16 + lm; if (mrow >= M) mrow = M - 1;
    const f16* __restrict__ arow = A + (size_t)mrow * K;

    f8v zero = {};
    f8v acc[NT];
#pragma unroll
    for (int i = 0; i < NT; ++i) acc[i] = zero;

    for (int kk = 0; kk < K; kk += KCH) {
        h16v af[KS];
#pragma unroll
        for (int k4 = 0; k4 < KS; ++k4) af[k4] = frag_load(arow + kk + k4 * 32, base0);
#pragma unroll
        for (int nt = 0; nt < NT; ++nt) {
            const f16* brow = &sW[(nt * 16 + lm) * LDSK + kk];
#pragma unroll
            for (int k4 = 0; k4 < KS; ++k4) {
                h16v b = frag_load(brow + k4 * 32, base0);
                acc[nt] = __builtin_amdgcn_wmma_f32_16x16x32_f16(
                    false, af[k4], false, b, (short)0, acc[nt], false, false);
            }
        }
    }

    // ---- epilogue ----
#pragma unroll
    for (int nt = 0; nt < NT; ++nt) {
        const int ncol = n0 + nt * 16 + lm;
        const float bn = (flags & 1) ? bias[ncol] : 0.0f;
#pragma unroll
        for (int r = 0; r < 8; ++r) {
            const int m = m0 + wv * 16 + hi * 8 + r;
            if (m >= M) continue;
            float v = acc[nt][r] + bn;
            if (flags & 2) v = silu_f(v);
            if (flags & 4) {
                const float* rb = rbf6 + (size_t)m * NR;
                float s = 0.0f;
#pragma unroll
                for (int k = 0; k < NR; ++k) s += rb[k] * R12[k * N + ncol];
                v *= s;
            }
            if (flags & 8) v += (float)addsrc[(size_t)m * N + ncol];
            if (flags & 16) outF[(size_t)m * N + ncol] += v;
            else            outH[(size_t)m * N + ncol] = (f16)v;
        }
    }
}

// ---------------- embedding gather-GEMM: x = silu([e_i | e_j | r] @ W(384x128) + b) ----------------
__global__ __launch_bounds__(32) void gemm_embed_kernel(
    const f16* __restrict__ embH, const f16* __restrict__ rH,
    const int* __restrict__ z, const int* __restrict__ ei, const int* __restrict__ ej,
    const f16* __restrict__ Wt, const float* __restrict__ bias,
    f16* __restrict__ outH, int M)
{
    const int N = 128, K = 384;
    const int tn = blockIdx.x % (N >> 4);
    const int tm = blockIdx.x / (N >> 4);
    const int m0 = tm << 4, n0 = tn << 4;
    const int lane = threadIdx.x, lm = lane & 15, hi = lane >> 4, base0 = hi * 8;

    int mrow = m0 + lm; if (mrow >= M) mrow = M - 1;
    const f16* src0 = embH + (size_t)z[ei[mrow]] * 128;
    const f16* src1 = embH + (size_t)z[ej[mrow]] * 128;
    const f16* src2 = rH + (size_t)mrow * 128;
    const int ncol = n0 + lm;
    const f16* __restrict__ brow = Wt + (size_t)ncol * K;

    f8v acc = {};
    for (int k0 = 0; k0 < K; k0 += 32) {
        const int seg = k0 >> 7;
        const f16* arow = (seg == 0 ? src0 : (seg == 1 ? src1 : src2)) + (k0 & 127);
        h16v a = frag_load(arow, base0);
        h16v b = frag_load(brow + k0, base0);
        acc = __builtin_amdgcn_wmma_f32_16x16x32_f16(false, a, false, b, (short)0, acc, false, false);
    }
    const float bn = bias[ncol];
#pragma unroll
    for (int r = 0; r < 8; ++r) {
        const int m = m0 + r + hi * 8;
        if (m >= M) continue;
        outH[(size_t)m * N + ncol] = (f16)silu_f(acc[r] + bn);
    }
}

// ---------------- small utility kernels ----------------
__global__ void fill0_kernel(float* __restrict__ p, size_t n) {
    size_t i = (size_t)blockIdx.x * blockDim.x + threadIdx.x;
    if (i < n) p[i] = 0.0f;
}

__global__ void f2h_kernel(const float* __restrict__ s, f16* __restrict__ d, size_t n) {
    size_t i = (size_t)blockIdx.x * blockDim.x + threadIdx.x;
    if (i < n) d[i] = (f16)s[i];
}

// W[K,N] f32 row-major -> Wt[N,K] f16
__global__ void wt_convert_kernel(const float* __restrict__ W, f16* __restrict__ Wt, int K, int N) {
    int idx = blockIdx.x * blockDim.x + threadIdx.x;
    if (idx >= K * N) return;
    int k = idx / N, n = idx % N;
    Wt[(size_t)n * K + k] = (f16)W[(size_t)k * N + n];
}

// R12[6,128] = rbf1[6,8] @ rbf2[8,128]
__global__ void r12_kernel(const float* __restrict__ r1, const float* __restrict__ r2, float* __restrict__ R12) {
    int idx = blockIdx.x * blockDim.x + threadIdx.x;
    if (idx >= NR * 128) return;
    int k = idx / 128, n = idx % 128;
    float s = 0.0f;
#pragma unroll
    for (int j = 0; j < 8; ++j) s += r1[k * 8 + j] * r2[j * 128 + n];
    R12[idx] = s;
}

// Ccomb[42,64] = sbf1[42,8] @ sbf2[8,64]
__global__ void ccomb_kernel(const float* __restrict__ s1, const float* __restrict__ s2, float* __restrict__ C) {
    int idx = blockIdx.x * blockDim.x + threadIdx.x;
    if (idx >= NS * NR * 64) return;
    int k = idx / 64, n = idx % 64;
    float s = 0.0f;
#pragma unroll
    for (int j = 0; j < 8; ++j) s += s1[k * 8 + j] * s2[j * 64 + n];
    C[idx] = s;
}

// per-edge distance + radial basis rbf[e][6]
__global__ void edge_geom_kernel(const float* __restrict__ pos,
                                 const int* __restrict__ ei, const int* __restrict__ ej,
                                 const float* __restrict__ freq,
                                 float* __restrict__ dist, float* __restrict__ rbf, int E)
{
    int e = blockIdx.x * blockDim.x + threadIdx.x;
    if (e >= E) return;
    int i = ei[e], j = ej[e];
    float dx = pos[3 * i] - pos[3 * j];
    float dy = pos[3 * i + 1] - pos[3 * j + 1];
    float dz = pos[3 * i + 2] - pos[3 * j + 2];
    float d = sqrtf(dx * dx + dy * dy + dz * dz);
    dist[e] = d;
    float x = d / CUTOFF_F;
    float env = envelope_f(x);
#pragma unroll
    for (int r = 0; r < NR; ++r) rbf[(size_t)e * NR + r] = env * __sinf(freq[r] * x);
}

// rH[e][ch] = silu(rbf[e] @ rbf_w + rbf_b)  (K=6)
__global__ void rh_kernel(const float* __restrict__ rbf, const float* __restrict__ w,
                          const float* __restrict__ b, f16* __restrict__ rH, int E)
{
    int idx = blockIdx.x * blockDim.x + threadIdx.x;
    if (idx >= E * 128) return;
    int e = idx / 128, ch = idx % 128;
    float acc = b[ch];
#pragma unroll
    for (int k = 0; k < NR; ++k) acc += rbf[(size_t)e * NR + k] * w[k * 128 + ch];
    rH[idx] = (f16)silu_f(acc);
}

// spherical bessel j_l (float)
__device__ __forceinline__ float sph_jn_f(int l, float x) {
    float j0 = __sinf(x) / x;
    if (l == 0) return j0;
    float jm = j0;
    float jc = __sinf(x) / (x * x) - __cosf(x) / x;
    for (int o = 1; o < l; ++o) { float t = (2.0f * o + 1.0f) / x * jc - jm; jm = jc; jc = t; }
    return jc;
}

__device__ double sph_jn_d(int l, double x) {
    double j0 = sin(x) / x;
    if (l == 0) return j0;
    double jm = j0;
    double jc = sin(x) / (x * x) - cos(x) / x;
    for (int o = 1; o < l; ++o) { double t = (2.0 * o + 1.0) / x * jc - jm; jm = jc; jc = t; }
    return jc;
}

// one-thread bisection for SBF_Z / SBF_NORM (module constants in the reference)
__global__ void init_consts_kernel(float* __restrict__ Zc, float* __restrict__ Nc) {
    if (threadIdx.x != 0 || blockIdx.x != 0) return;
    const double PI = 3.14159265358979323846;
    double pts[NS + NR - 1];
    double zr[NS][NR];
    int len = NS + NR - 1;  // 12
    for (int i = 0; i < len; ++i) pts[i] = (i + 1) * PI;
    for (int r = 0; r < NR; ++r) zr[0][r] = pts[r];
    for (int l = 1; l < NS; ++l) {
        double a[NS + NR - 1], b[NS + NR - 1], fa[NS + NR - 1];
        int m = len - 1;
        for (int i = 0; i < m; ++i) { a[i] = pts[i]; b[i] = pts[i + 1]; fa[i] = sph_jn_d(l, a[i]); }
        for (int it = 0; it < 80; ++it) {
            for (int i = 0; i < m; ++i) {
                double mid = 0.5 * (a[i] + b[i]);
                double fm = sph_jn_d(l, mid);
                if (fa[i] * fm <= 0.0) { b[i] = mid; }
                else { a[i] = mid; fa[i] = fm; }
            }
        }
        for (int i = 0; i < m; ++i) pts[i] = 0.5 * (a[i] + b[i]);
        len = m;
        for (int r = 0; r < NR; ++r) zr[l][r] = pts[r];
    }
    for (int l = 0; l < NS; ++l)
        for (int r = 0; r < NR; ++r) {
            Zc[l * NR + r] = (float)zr[l][r];
            double jv = sph_jn_d(l + 1, zr[l][r]);
            Nc[l * NR + r] = (float)(1.0 / sqrt(0.5 * jv * jv));
        }
}

// per-triplet spherical basis sbf[t][42]
__global__ void sbf_kernel(const float* __restrict__ pos,
                           const int* __restrict__ ii, const int* __restrict__ ij, const int* __restrict__ ik,
                           const int* __restrict__ ikj, const float* __restrict__ dist,
                           const float* __restrict__ Zc, const float* __restrict__ Nc,
                           f16* __restrict__ sbf, int T)
{
    int t = blockIdx.x * blockDim.x + threadIdx.x;
    if (t >= T) return;
    int i = ii[t], j = ij[t], k = ik[t];
    float jx = pos[3 * j] - pos[3 * i], jy = pos[3 * j + 1] - pos[3 * i + 1], jz = pos[3 * j + 2] - pos[3 * i + 2];
    float kx = pos[3 * k] - pos[3 * i], ky = pos[3 * k + 1] - pos[3 * i + 1], kz = pos[3 * k + 2] - pos[3 * i + 2];
    float a = jx * kx + jy * ky + jz * kz;
    float cx = jy * kz - jz * ky, cy = jz * kx - jx * kz, cz = jx * ky - jy * kx;
    float b = sqrtf(cx * cx + cy * cy + cz * cz);
    float angle = atan2f(b, a);
    float c = __cosf(angle);

    float pls[NS];
    pls[0] = 1.0f; pls[1] = c;
    for (int l = 1; l < NS - 1; ++l)
        pls[l + 1] = ((2.0f * l + 1.0f) * c * pls[l] - (float)l * pls[l - 1]) / (float)(l + 1);

    float xd = dist[ikj[t]] / CUTOFF_F;
    float env = envelope_f(xd);
    const float INV4PI = 0.07957747154594767f;  // 1/(4*pi)
#pragma unroll
    for (int l = 0; l < NS; ++l) {
        float coef = sqrtf((2.0f * l + 1.0f) * INV4PI) * pls[l];
#pragma unroll
        for (int r = 0; r < NR; ++r) {
            float rad = Nc[l * NR + r] * sph_jn_f(l, Zc[l * NR + r] * xd) * env;
            sbf[(size_t)t * (NS * NR) + l * NR + r] = (f16)(rad * coef);
        }
    }
}

// lower_bound of e in sorted idx_ji
__global__ void estart_kernel(const int* __restrict__ idx_ji, int* __restrict__ es, int E, int T) {
    int e = blockIdx.x * blockDim.x + threadIdx.x;
    if (e > E) return;
    int lo = 0, hi = T;
    while (lo < hi) { int mid = (lo + hi) >> 1; if (idx_ji[mid] < e) lo = mid + 1; else hi = mid; }
    es[e] = lo;
}

// per-edge triplet aggregation: out[e][ch] = sum_t (sbf[t]@Ccomb)[ch] * xkj2[idx_kj[t]][ch]
__global__ __launch_bounds__(64) void aggregate_kernel(
    const f16* __restrict__ xkj2, const f16* __restrict__ sbf,
    const float* __restrict__ Ccomb, const int* __restrict__ ikj,
    const int* __restrict__ es, f16* __restrict__ out, int E)
{
    int e = blockIdx.x;
    int ch = threadIdx.x;
    __shared__ float ss[NS * NR];
    int t0 = es[e], t1 = es[e + 1];
    float acc = 0.0f;
    for (int t = t0; t < t1; ++t) {
        if (ch < NS * NR) ss[ch] = (float)sbf[(size_t)t * (NS * NR) + ch];
        __syncthreads();
        float s = 0.0f;
#pragma unroll
        for (int k = 0; k < NS * NR; ++k) s += ss[k] * Ccomb[k * 64 + ch];
        acc += s * (float)xkj2[(size_t)ikj[t] * 64 + ch];
        __syncthreads();
    }
    out[(size_t)e * 64 + ch] = (f16)acc;
}

// g[e] = (rbf[e]@prbf) * x[e]; atomic scatter into gnode[edge_i[e]]
__global__ void outscatter_kernel(const float* __restrict__ rbf, const float* __restrict__ prbf,
                                  const f16* __restrict__ x, const int* __restrict__ ei,
                                  float* __restrict__ gnode, int E)
{
    int e = blockIdx.x;
    int ch = threadIdx.x;  // 128
    if (e >= E) return;
    float r = 0.0f;
#pragma unroll
    for (int k = 0; k < NR; ++k) r += rbf[(size_t)e * NR + k] * prbf[k * 128 + ch];
    float v = r * (float)x[(size_t)e * 128 + ch];
    atomicAdd(&gnode[(size_t)ei[e] * 128 + ch], v);
}

// per-graph reduction + leaky_relu + head
__global__ void head_kernel(const float* __restrict__ P, const float* __restrict__ hw,
                            const float* __restrict__ hb, float* __restrict__ out, int apg)
{
    __shared__ float red[16];
    int g = blockIdx.x;
    int ch = threadIdx.x;  // 16
    float s = 0.0f;
    for (int a = 0; a < apg; ++a) s += P[((size_t)g * apg + a) * 16 + ch];
    s = (s > 0.0f) ? s : 0.01f * s;
    red[ch] = s * hw[ch];
    __syncthreads();
    if (ch == 0) {
        float t = hb[0];
        for (int i = 0; i < 16; ++i) t += red[i];
        out[g] = t;
    }
}

// ---------------- host-side GEMM dispatch ----------------
static void launch_gemm(hipStream_t s, const f16* A, const f16* Wt, const float* bias,
                        const f16* add, const float* rb, const float* r12,
                        f16* oH, float* oF, int M, int K, int N, int flags)
{
    const int NCH = (N >= 128) ? 128 : N;
    const unsigned grid = (unsigned)(((size_t)((M + 63) / 64)) * (N / NCH));
    if (K == 128 && NCH == 128)
        gemm_lds_kernel<8, 128><<<grid, 128, 0, s>>>(A, Wt, bias, add, rb, r12, oH, oF, M, N, flags);
    else if (K == 128 && NCH == 64)
        gemm_lds_kernel<4, 128><<<grid, 128, 0, s>>>(A, Wt, bias, add, rb, r12, oH, oF, M, N, flags);
    else if (K == 64)
        gemm_lds_kernel<8, 64><<<grid, 128, 0, s>>>(A, Wt, bias, add, rb, r12, oH, oF, M, N, flags);
    else if (K == 256 && NCH == 128)
        gemm_lds_kernel<8, 256><<<grid, 128, 0, s>>>(A, Wt, bias, add, rb, r12, oH, oF, M, N, flags);
    else  // K == 256, N == 16
        gemm_lds_kernel<1, 256><<<grid, 128, 0, s>>>(A, Wt, bias, add, rb, r12, oH, oF, M, N, flags);
}

// ---------------- host orchestration ----------------
extern "C" void kernel_launch(void* const* d_in, const int* in_sizes, int n_in,
                              void* d_out, int out_size, void* d_ws, size_t ws_size,
                              hipStream_t stream)
{
    (void)n_in; (void)ws_size;
    const int* z      = (const int*)d_in[0];
    const float* pos  = (const float*)d_in[1];
    const int* edge_j = (const int*)d_in[3];
    const int* edge_i = (const int*)d_in[4];
    const int* idx_i  = (const int*)d_in[5];
    const int* idx_jj = (const int*)d_in[6];
    const int* idx_k  = (const int*)d_in[7];
    const int* idx_kj = (const int*)d_in[8];
    const int* idx_ji = (const int*)d_in[9];
    const float* freq = (const float*)d_in[10];

    const int NATOMS = in_sizes[0];
    const int E = in_sizes[3];
    const int T = in_sizes[5];
    const int GRAPHS = out_size;
    const int APG = NATOMS / GRAPHS;

    auto PF = [&](int i) { return (const float*)d_in[i]; };

    // ---- workspace layout ----
    char* base = (char*)d_ws;
    size_t off = 0;
    auto alloc = [&](size_t bytes) -> char* {
        char* p = base + off;
        off = (off + bytes + 255) & ~(size_t)255;
        return p;
    };
    float* Zc    = (float*)alloc(NS * NR * sizeof(float));
    float* Nc    = (float*)alloc(NS * NR * sizeof(float));
    float* distD = (float*)alloc((size_t)E * sizeof(float));
    float* rbfD  = (float*)alloc((size_t)E * NR * sizeof(float));
    f16*   rHD   = (f16*)alloc((size_t)E * 128 * sizeof(f16));
    f16*   sbfD  = (f16*)alloc((size_t)T * NS * NR * sizeof(f16));
    int*   esD   = (int*)alloc((size_t)(E + 1) * sizeof(int));
    f16*   B0    = (f16*)alloc((size_t)E * 128 * sizeof(f16));  // x
    f16*   B1    = (f16*)alloc((size_t)E * 128 * sizeof(f16));
    f16*   B2    = (f16*)alloc((size_t)E * 128 * sizeof(f16));
    f16*   B6    = (f16*)alloc((size_t)E * 128 * sizeof(f16));
    f16*   B7    = (f16*)alloc((size_t)E * 128 * sizeof(f16));
    f16*   B4    = (f16*)alloc((size_t)E * 64 * sizeof(f16));
    f16*   B5    = (f16*)alloc((size_t)E * 64 * sizeof(f16));
    float* gnode = (float*)alloc((size_t)NATOMS * 128 * sizeof(float));
    f16*   gnodeH= (f16*)alloc((size_t)NATOMS * 128 * sizeof(f16));
    f16*   G1    = (f16*)alloc((size_t)NATOMS * 256 * sizeof(f16));
    f16*   G2    = (f16*)alloc((size_t)NATOMS * 256 * sizeof(f16));
    float* Pacc  = (float*)alloc((size_t)NATOMS * 16 * sizeof(float));
    float* R12   = (float*)alloc((size_t)NR * 128 * sizeof(float));
    float* Ccomb = (float*)alloc((size_t)NS * NR * 64 * sizeof(float));
    f16*   embH  = (f16*)alloc((size_t)95 * 128 * sizeof(f16));
    f16*   wslot = (f16*)alloc((size_t)256 * 384 * sizeof(f16));  // covers up to 256x256 and 384x128

    f16* pX  = B0;
    f16* pB7 = B7;

    auto fill = [&](float* p, size_t n) {
        fill0_kernel<<<(unsigned)((n + 255) / 256), 256, 0, stream>>>(p, n);
    };
    auto dense = [&](const float* W, int K, int N, const f16* A, const float* bias,
                     const f16* add, const float* rb, const float* r12,
                     f16* oH, float* oF, int M, int flags) {
        wt_convert_kernel<<<(unsigned)(((size_t)K * N + 255) / 256), 256, 0, stream>>>(W, wslot, K, N);
        launch_gemm(stream, A, wslot, bias, add, rb, r12, oH, oF, M, K, N, flags);
    };
    auto outblock = [&](int ob) {
        int bidx = 112 + ob * 9;
        fill(gnode, (size_t)NATOMS * 128);
        outscatter_kernel<<<E, 128, 0, stream>>>(rbfD, PF(bidx + 0), pX, edge_i, gnode, E);
        f2h_kernel<<<(unsigned)(((size_t)NATOMS * 128 + 255) / 256), 256, 0, stream>>>(gnode, gnodeH, (size_t)NATOMS * 128);
        dense(PF(bidx + 1), 128, 256, gnodeH, nullptr, nullptr, nullptr, nullptr, G1, nullptr, NATOMS, 0);
        dense(PF(bidx + 2), 256, 256, G1, PF(bidx + 3), nullptr, nullptr, nullptr, G2, nullptr, NATOMS, 1 | 2);
        dense(PF(bidx + 4), 256, 256, G2, PF(bidx + 5), nullptr, nullptr, nullptr, G1, nullptr, NATOMS, 1 | 2);
        dense(PF(bidx + 6), 256, 256, G1, PF(bidx + 7), nullptr, nullptr, nullptr, G2, nullptr, NATOMS, 1 | 2);
        dense(PF(bidx + 8), 256, 16, G2, nullptr, nullptr, nullptr, nullptr, nullptr, Pacc, NATOMS, 16);
    };

    // ---- preprocessing ----
    init_consts_kernel<<<1, 1, 0, stream>>>(Zc, Nc);
    f2h_kernel<<<(95 * 128 + 255) / 256, 256, 0, stream>>>(PF(11), embH, 95 * 128);
    edge_geom_kernel<<<(E + 255) / 256, 256, 0, stream>>>(pos, edge_i, edge_j, freq, distD, rbfD, E);
    rh_kernel<<<(unsigned)(((size_t)E * 128 + 255) / 256), 256, 0, stream>>>(rbfD, PF(12), PF(13), rHD, E);
    estart_kernel<<<(E + 1 + 255) / 256, 256, 0, stream>>>(idx_ji, esD, E, T);
    sbf_kernel<<<(T + 255) / 256, 256, 0, stream>>>(pos, idx_i, idx_jj, idx_k, idx_kj, distD, Zc, Nc, sbfD, T);

    // ---- embedding: x = silu([e_i|e_j|r] @ W + b) ----
    wt_convert_kernel<<<(384 * 128 + 255) / 256, 256, 0, stream>>>(PF(14), wslot, 384, 128);
    {
        unsigned tiles = (unsigned)(((E + 15) / 16) * 8);
        gemm_embed_kernel<<<tiles, 32, 0, stream>>>(embH, rHD, z, edge_i, edge_j, wslot, PF(15), pX, E);
    }

    fill(Pacc, (size_t)NATOMS * 16);
    outblock(0);

    // ---- interactions ----
    for (int ib = 0; ib < 4; ++ib) {
        int b = 16 + ib * 24;
        r12_kernel<<<3, 256, 0, stream>>>(PF(b + 0), PF(b + 1), R12);
        ccomb_kernel<<<11, 256, 0, stream>>>(PF(b + 2), PF(b + 3), Ccomb);

        // x_ji = silu(x@ji_w + ji_b)
        dense(PF(b + 6), 128, 128, pX, PF(b + 7), nullptr, nullptr, nullptr, B1, nullptr, E, 1 | 2);
        // x_kj = silu(x@kj_w + kj_b) * (rbf@rbf1@rbf2)
        dense(PF(b + 4), 128, 128, pX, PF(b + 5), nullptr, rbfD, R12, B2, nullptr, E, 1 | 2 | 4);
        // x_kj2 = silu(x_kj @ down)
        dense(PF(b + 8), 128, 64, B2, nullptr, nullptr, nullptr, nullptr, B4, nullptr, E, 2);
        // segment-sum over triplets
        aggregate_kernel<<<E, 64, 0, stream>>>(B4, sbfD, Ccomb, idx_kj, esD, B5, E);
        // h = silu(agg @ up) + x_ji
        dense(PF(b + 9), 64, 128, B5, nullptr, B1, nullptr, nullptr, B6, nullptr, E, 2 | 8);
        // before residual
        dense(PF(b + 10), 128, 128, B6, PF(b + 11), nullptr, nullptr, nullptr, pB7, nullptr, E, 1 | 2);
        dense(PF(b + 12), 128, 128, pB7, PF(b + 13), B6, nullptr, nullptr, B6, nullptr, E, 1 | 2 | 8);
        // h = silu(h@lin + b) + x
        dense(PF(b + 14), 128, 128, B6, PF(b + 15), pX, nullptr, nullptr, pB7, nullptr, E, 1 | 2 | 8);
        // after residuals (x2)
        dense(PF(b + 16), 128, 128, pB7, PF(b + 17), nullptr, nullptr, nullptr, B1, nullptr, E, 1 | 2);
        dense(PF(b + 18), 128, 128, B1, PF(b + 19), pB7, nullptr, nullptr, pB7, nullptr, E, 1 | 2 | 8);
        dense(PF(b + 20), 128, 128, pB7, PF(b + 21), nullptr, nullptr, nullptr, B1, nullptr, E, 1 | 2);
        dense(PF(b + 22), 128, 128, B1, PF(b + 23), pB7, nullptr, nullptr, pB7, nullptr, E, 1 | 2 | 8);

        // x <- h
        f16* tmp = pX; pX = pB7; pB7 = tmp;
        outblock(ib + 1);
    }

    // ---- head ----
    head_kernel<<<GRAPHS, 16, 0, stream>>>(Pacc, PF(157), PF(158), (float*)d_out, APG);
}
